// KipfNet_simple_30210799960803
// MI455X (gfx1250) — matrix operand order repaired and checked by reading, not compiled, not run on hardware
//
#include <hip/hip_runtime.h>

#define N_NODES 50000
#define N_EDGES 800000
#define F_IN 128
#define F_OUT 64
#define K_ORD 8

typedef __attribute__((ext_vector_type(16))) _Float16 v16h;
typedef __attribute__((ext_vector_type(8)))  float    v8f;

// ---------------- elementwise helpers (float4-vectorized) ----------------

__global__ void zero_f32(float* __restrict__ p, long n) {
  long i = (long)blockIdx.x * blockDim.x + threadIdx.x;
  long stride = (long)gridDim.x * blockDim.x;
  for (; i < n; i += stride) p[i] = 0.0f;
}

__global__ void zero_f32_v4(float4* __restrict__ p, long n4) {
  long i = (long)blockIdx.x * blockDim.x + threadIdx.x;
  long stride = (long)gridDim.x * blockDim.x;
  float4 z = {0.0f, 0.0f, 0.0f, 0.0f};
  for (; i < n4; i += stride) p[i] = z;
}

// o = -a  (init for fused Chebyshev recurrence)
__global__ void neg_copy_v4(const float4* __restrict__ a, float4* __restrict__ o, long n4) {
  long i = (long)blockIdx.x * blockDim.x + threadIdx.x;
  long stride = (long)gridDim.x * blockDim.x;
  for (; i < n4; i += stride) {
    float4 v = a[i];
    float4 r = {-v.x, -v.y, -v.z, -v.w};
    o[i] = r;
  }
}

// deg[src] += 1  (float segment_sum of ones; exact below 2^24)
__global__ void deg_kernel(const int* __restrict__ ei, float* __restrict__ deg, int E) {
  int e = blockIdx.x * blockDim.x + threadIdx.x;
  if (e < E) unsafeAtomicAdd(&deg[ei[e]], 1.0f);
}

// deg -> d^{-1/2} in place
__global__ void dis_kernel(float* __restrict__ deg, int n) {
  int i = blockIdx.x * blockDim.x + threadIdx.x;
  if (i < n) { float d = deg[i]; deg[i] = (d > 0.0f) ? rsqrtf(d) : 0.0f; }
}

// w[e] = -dis[src] * dis[dst]
__global__ void edgew_kernel(const int* __restrict__ ei, const float* __restrict__ dis,
                             float* __restrict__ wE, int E) {
  int e = blockIdx.x * blockDim.x + threadIdx.x;
  if (e < E) wE[e] = -dis[ei[e]] * dis[ei[E + e]];
}

// ---------------- W pre-split: f32 [k][kk][n] -> f16 hi/lo transposed [k][n][kk] ----------------
// Makes each lane's B fragment (K = 16*half + e, e contiguous) one 32-byte vector load.

__global__ void split_w(const float* __restrict__ W, _Float16* __restrict__ Whi,
                        _Float16* __restrict__ Wlo, int total) {
  int i = blockIdx.x * blockDim.x + threadIdx.x;
  if (i >= total) return;
  int k  = i >> 13;        // / (128*64)
  int r  = i & 8191;
  int kk = r >> 6;         // K index 0..127
  int n  = r & 63;         // col  0..63
  float v = W[i];
  _Float16 h = (_Float16)v;
  int o = (k << 13) + (n << 7) + kk;   // [k][n][kk]
  Whi[o] = h;
  Wlo[o] = (_Float16)(v - (float)h);
}

// ---------------- scatter SpMM: tout[dst] += scale * w[e] * tin[src] ----------------
// One wave per edge; 32 lanes x float4 = 128 features. Edge metadata is wave-uniform:
// route the wave index through readfirstlane so src/dst/w come in as scalar (SMEM) loads.

__global__ void spmm_scatter(const int* __restrict__ ei, const float* __restrict__ wE,
                             const float* __restrict__ tin, float* __restrict__ tout,
                             float scale, int E) {
  const int wv = __builtin_amdgcn_readfirstlane((int)(threadIdx.x >> 5));
  const int gwave = blockIdx.x * 8 + wv;          // blockDim.x == 256 -> 8 waves
  const int lane = threadIdx.x & 31;
  if (gwave >= E) return;                          // scalar branch; never taken (exact grid)
  const int s = ei[gwave];                         // s_load (uniform address)
  const int d = ei[E + gwave];                     // s_load
  const float we = wE[gwave] * scale;              // s_load + v_mul
  float4 v = *((const float4*)(tin + (long)s * F_IN) + lane);
  float* o = tout + (long)d * F_IN + (long)lane * 4;
  unsafeAtomicAdd(o + 0, we * v.x);
  unsafeAtomicAdd(o + 1, we * v.y);
  unsafeAtomicAdd(o + 2, we * v.z);
  unsafeAtomicAdd(o + 3, we * v.w);
}

// ---------------- WMMA GEMM: acc[50000x64] += T[50000x128] @ Wk[128x64] ----------------
// One wave per 16-row strip; A fragments built once per K-chunk and reused across all
// 4 column tiles (4 v8f accumulators). B fragments are preconverted f16 hi/lo, loaded
// as contiguous 32B vectors. Split-f16 3-pass: A*B ~= Ah*Bh + Al*Bh + Ah*Bl.

__global__ void __launch_bounds__(32)
wmma_gemm_acc(const float* __restrict__ T, const _Float16* __restrict__ Whik,
              const _Float16* __restrict__ Wlok, float* __restrict__ acc) {
  const int lane = threadIdx.x & 31;
  const int half = lane >> 4;            // 0 / 1
  const int l16  = lane & 15;
  const long row = (long)blockIdx.x * 16 + l16;   // A row this lane feeds
  const int crow0 = blockIdx.x * 16 + half * 8;   // C: VGPR r holds row crow0+r, col = lane%16

  v8f c[4];
#pragma unroll
  for (int ct = 0; ct < 4; ++ct)
#pragma unroll
    for (int r = 0; r < 8; ++r)
      c[ct][r] = acc[(long)(crow0 + r) * F_OUT + ct * 16 + l16];

#pragma unroll
  for (int kc = 0; kc < F_IN / 32; ++kc) {
    // A 16x32 f16 fragment: lane row = l16; elem e -> K = kc*32 + half*16 + e (e = 0..15),
    // one 64B contiguous run (matches B's K order).
    const float* ap = T + row * F_IN + kc * 32 + half * 16;
    v16h ah, al;
#pragma unroll
    for (int e = 0; e < 8; ++e) {
      float v0 = ap[e];
      _Float16 h0 = (_Float16)v0;
      ah[e] = h0;
      al[e] = (_Float16)(v0 - (float)h0);
      float v1 = ap[8 + e];
      _Float16 h1 = (_Float16)v1;
      ah[8 + e] = h1;
      al[8 + e] = (_Float16)(v1 - (float)h1);
    }
#pragma unroll
    for (int ct = 0; ct < 4; ++ct) {
      // B fragment: col = ct*16 + l16, K = kc*32 + half*16 + e -> contiguous in Wt[col][K]
      const long bo = (long)(ct * 16 + l16) * F_IN + kc * 32 + half * 16;
      v16h bh = *(const v16h*)(Whik + bo);
      v16h bl = *(const v16h*)(Wlok + bo);
      c[ct] = __builtin_amdgcn_wmma_f32_16x16x32_f16(false, ah, false, bh, (short)0, c[ct], false, false);
      c[ct] = __builtin_amdgcn_wmma_f32_16x16x32_f16(false, al, false, bh, (short)0, c[ct], false, false);
      c[ct] = __builtin_amdgcn_wmma_f32_16x16x32_f16(false, ah, false, bl, (short)0, c[ct], false, false);
    }
  }

#pragma unroll
  for (int ct = 0; ct < 4; ++ct)
#pragma unroll
    for (int r = 0; r < 8; ++r)
      acc[(long)(crow0 + r) * F_OUT + ct * 16 + l16] = c[ct][r];
}

// ---------------- out = relu(out + b) ----------------

__global__ void bias_relu(float* __restrict__ out, const float* __restrict__ b, long n) {
  long i = (long)blockIdx.x * blockDim.x + threadIdx.x;
  if (i < n) {
    float v = out[i] + b[i & (F_OUT - 1)];
    out[i] = v > 0.0f ? v : 0.0f;
  }
}

// ---------------- driver ----------------

extern "C" void kernel_launch(void* const* d_in, const int* in_sizes, int n_in,
                              void* d_out, int out_size, void* d_ws, size_t ws_size,
                              hipStream_t stream) {
  (void)in_sizes; (void)n_in; (void)out_size; (void)ws_size;
  const float* x  = (const float*)d_in[0];
  const int*   ei = (const int*)d_in[1];   // [2][E] int32
  const float* W  = (const float*)d_in[2]; // [8][128][64]
  const float* b  = (const float*)d_in[3]; // [64]
  float* out = (float*)d_out;              // [50000][64]

  float* dis = (float*)d_ws;               // N
  float* wE  = dis + N_NODES;              // E
  float* B1  = wE + N_EDGES;               // N*128 each
  float* B2  = B1 + (long)N_NODES * F_IN;
  float* B3  = B2 + (long)N_NODES * F_IN;
  _Float16* Whi = (_Float16*)(B3 + (long)N_NODES * F_IN);  // 8*64*128 halves (32B aligned)
  _Float16* Wlo = Whi + (long)K_ORD * F_IN * F_OUT;

  const int TB = 256;
  const int gN  = (N_NODES + TB - 1) / TB;
  const int gE  = (N_EDGES + TB - 1) / TB;
  const int gSp = N_EDGES / (TB / 32);     // one wave per edge: 100000 blocks exactly
  const int gG  = N_NODES / 16;            // 3125 row strips exactly
  const long nT4 = (long)N_NODES * F_IN / 4;   // float4 count of one T buffer

  // fresh state every call (graph-replay safe)
  zero_f32<<<gN, TB, 0, stream>>>(dis, N_NODES);
  zero_f32<<<2048, TB, 0, stream>>>(out, (long)N_NODES * F_OUT);
  zero_f32_v4<<<2048, TB, 0, stream>>>((float4*)B1, nT4);

  split_w<<<(K_ORD * F_IN * F_OUT) / TB, TB, 0, stream>>>(W, Whi, Wlo, K_ORD * F_IN * F_OUT);

  deg_kernel<<<gE, TB, 0, stream>>>(ei, dis, N_EDGES);
  dis_kernel<<<gN, TB, 0, stream>>>(dis, N_NODES);
  edgew_kernel<<<gE, TB, 0, stream>>>(ei, dis, wE, N_EDGES);

  // T1 = prop(x);  out += T0 @ W0 + T1 @ W1
  spmm_scatter<<<gSp, TB, 0, stream>>>(ei, wE, x, B1, 1.0f, N_EDGES);
  wmma_gemm_acc<<<gG, 32, 0, stream>>>(x,  Whi + 0 * F_IN * F_OUT, Wlo + 0 * F_IN * F_OUT, out);
  wmma_gemm_acc<<<gG, 32, 0, stream>>>(B1, Whi + 1 * F_IN * F_OUT, Wlo + 1 * F_IN * F_OUT, out);

  // T_k = 2*prop(T_{k-1}) - T_{k-2}, fused as: Tnext = -Tprev; Tnext += 2*w*Tcur[src]
  const float* Tprev = x;
  const float* Tcur  = B1;
  float* rot[3] = {B2, B3, B1};
  for (int k = 2; k < K_ORD; ++k) {
    float* Tnext = rot[(k - 2) % 3];
    neg_copy_v4<<<2048, TB, 0, stream>>>((const float4*)Tprev, (float4*)Tnext, nT4);
    spmm_scatter<<<gSp, TB, 0, stream>>>(ei, wE, Tcur, Tnext, 2.0f, N_EDGES);
    wmma_gemm_acc<<<gG, 32, 0, stream>>>(Tnext, Whi + k * F_IN * F_OUT, Wlo + k * F_IN * F_OUT, out);
    Tprev = Tcur;
    Tcur  = Tnext;
  }

  bias_relu<<<((long)N_NODES * F_OUT + TB - 1) / TB, TB, 0, stream>>>(out, b, (long)N_NODES * F_OUT);
}
